// BiasedMHABlock_32444182954179
// MI455X (gfx1250) — compile-verified, matrix operand order checked by reading, not run on hardware
//
#include <hip/hip_runtime.h>
#include <hip/hip_bf16.h>

// ---------------- problem constants ----------------
#define BATCH   8
#define ULEN    1024
#define DMODEL  512
#define NHEAD   8
#define DHEAD   64
#define DFF     4096
#define RELMAX  128
#define NROWS   (BATCH * ULEN)          // 8192

// ---------------- WMMA vector types ----------------
typedef __attribute__((ext_vector_type(16))) __bf16        v16bf;
typedef __attribute__((ext_vector_type(8)))  float         v8f;
typedef __attribute__((ext_vector_type(4)))  unsigned int  u32x4;
typedef __attribute__((ext_vector_type(8)))  int           i32x8;
typedef __attribute__((ext_vector_type(4)))  int           i32x4;

// ---------------- feature probes ----------------
#if __has_builtin(__builtin_amdgcn_tensor_load_to_lds)
#define HAVE_TDM 1
#else
#define HAVE_TDM 0
#endif

#if __has_builtin(__builtin_amdgcn_s_wait_asynccnt)
#define WAIT_ASYNC(n) __builtin_amdgcn_s_wait_asynccnt(n)
#else
#define WAIT_ASYNC(n) asm volatile("s_wait_asynccnt %0" :: "i"(n) : "memory")
#endif

#if __has_builtin(__builtin_amdgcn_s_wait_tensorcnt)
#define WAIT_TENSOR(n) __builtin_amdgcn_s_wait_tensorcnt(n)
#else
#define WAIT_TENSOR(n) asm volatile("s_wait_tensorcnt %0" :: "i"(n) : "memory")
#endif

__device__ __forceinline__ unsigned short f2bf(float f) {
    unsigned int u = __float_as_uint(f);
    unsigned int r = u + 0x7FFFu + ((u >> 16) & 1u);   // round-to-nearest-even
    return (unsigned short)(r >> 16);
}
__device__ __forceinline__ float bf2f(unsigned short h) {
    return __uint_as_float(((unsigned int)h) << 16);
}

// 16B async copy global -> LDS (ASYNCcnt).  ISA 15.18: vdst = LDS byte addr,
// vaddr = 64-bit global address.  Generic LDS pointer low 32 bits == LDS offset.
__device__ __forceinline__ void async_ld16(const unsigned short* g, unsigned short* l) {
    unsigned int       loff = (unsigned int)(unsigned long long)l;
    unsigned long long ga   = (unsigned long long)g;
    asm volatile("global_load_async_to_lds_b128 %0, %1, off"
                 :: "v"(loff), "v"(ga) : "memory");
}

#if HAVE_TDM
// TDM: DMA a (128 rows x 32 cols) bf16 tile (row stride K elems) into LDS with
// hardware padding of 16B after every 64B -> LDS row stride 40 ushorts.
__device__ __forceinline__ void tdm_load_b_tile(const unsigned short* gsrc,
                                                unsigned short* ldst, int K) {
    unsigned long long ga  = (unsigned long long)gsrc;
    unsigned int       lds = (unsigned int)(unsigned long long)ldst;
    u32x4 g0;
    g0[0] = 1u;                                   // count=1, is_restore=0
    g0[1] = lds;                                  // lds_addr
    g0[2] = (unsigned int)(ga & 0xffffffffu);     // global_addr[31:0]
    g0[3] = (unsigned int)((ga >> 32) & 0x1ffffffu) | (2u << 30); // addr[56:32]|type=2
    const unsigned int td1 = 1u << 20;            // generous tensor_dim1 (no OOB)
    i32x8 g1;
    g1[0] = (int)((1u << 16)            // data_size = 2 bytes
                | (1u << 20)            // pad_enable
                | (3u << 22)            // pad_interval: 64B
                | (3u << 25));          // pad_amount: 16B
    g1[1] = (int)(((unsigned int)K & 0xffffu) << 16);               // tensor_dim0 lo
    g1[2] = (int)((((unsigned int)K >> 16) & 0xffffu)
                | ((td1 & 0xffffu) << 16));                          // dim0 hi | dim1 lo
    g1[3] = (int)(((td1 >> 16) & 0xffffu) | (32u << 16));            // dim1 hi | tile_dim0=32
    g1[4] = 128;                                                     // tile_dim1=128
    g1[5] = (int)K;                                                  // dim0_stride lo32
    g1[6] = 0;
    g1[7] = 0;
    i32x4 z4 = {0, 0, 0, 0};
#if __clang_major__ >= 23
    i32x8 z8 = {0, 0, 0, 0, 0, 0, 0, 0};
    __builtin_amdgcn_tensor_load_to_lds(g0, g1, z4, z4, z8, 0);
#else
    __builtin_amdgcn_tensor_load_to_lds(g0, g1, z4, z4, 0);
#endif
}
#endif

// Load a 16x32 bf16 A/B fragment from a row-major tile (stride in elements).
// ISA 7.12.2: lanes 0-15 -> rows, K {0..7,16..23}; lanes 16-31 -> K {8..15,24..31}.
__device__ __forceinline__ v16bf ldfrag(const unsigned short* __restrict__ base,
                                        int stride, int lane) {
    const int half = lane >> 4;
    const int r    = lane & 15;
    const unsigned short* p = base + (size_t)r * stride + half * 8;
    union { v16bf bf; uint4 q[2]; } u;
    u.q[0] = *(const uint4*)(p);
    u.q[1] = *(const uint4*)(p + 16);
    return u.bf;
}

__device__ __forceinline__ v8f wmma_bf16(v16bf a, v16bf b, v8f c) {
    return __builtin_amdgcn_wmma_f32_16x16x32_bf16(false, a, false, b,
                                                   (short)0, c, false, false);
}

// ---------------- prep: X -> bf16, Xn -> bf16 (row-normalized) ----------------
__global__ __launch_bounds__(256) void prep_x_kernel(
    const float* __restrict__ X, unsigned short* __restrict__ Xbf,
    unsigned short* __restrict__ Xnbf) {
    const int row = blockIdx.x;
    const int tid = threadIdx.x;
    __shared__ float red[256];
    const float* xr = X + (size_t)row * DMODEL;
    float a = xr[tid], b = xr[tid + 256];
    red[tid] = a * a + b * b;
    __syncthreads();
    for (int o = 128; o > 0; o >>= 1) {
        if (tid < o) red[tid] += red[tid + o];
        __syncthreads();
    }
    float inv = 1.0f / fmaxf(sqrtf(red[0]), 1e-6f);
    unsigned short* xo  = Xbf  + (size_t)row * DMODEL;
    unsigned short* xno = Xnbf + (size_t)row * DMODEL;
    xo[tid]        = f2bf(a);
    xo[tid + 256]  = f2bf(b);
    xno[tid]       = f2bf(a * inv);
    xno[tid + 256] = f2bf(b * inv);
}

// ---------------- weight transpose: W (KxN f32) -> Wt (NxK bf16) ----------------
__global__ void wtrans_kernel(const float* __restrict__ W,
                              unsigned short* __restrict__ Wt, int K, int N) {
    int n = blockIdx.x * 16 + threadIdx.x;
    int k = blockIdx.y * 16 + threadIdx.y;
    if (n < N && k < K)
        Wt[(size_t)n * K + k] = f2bf(W[(size_t)k * N + n]);
}

// ---------------- generic bf16 WMMA GEMM: C = A(MxK) * Bt(NxK)^T ----------------
// Double-buffered LDS tiles: A via per-lane async-to-LDS, B via TDM (fallback async).
// Steady-state loop is branch-free: issue(next) -> wait(<=1) -> barrier -> 4xWMMA.
// out_mode: 0 = f32 row-major (+opt bf16)   1 = QKV scatter [b][h][u][dh]
//           2 = V^T scatter [b][h][dh][u]   3 = bf16 row-major
__global__ __launch_bounds__(256) void gemm_bf16_kernel(
    const unsigned short* __restrict__ A, const unsigned short* __restrict__ Bt,
    const float* __restrict__ bias, float* __restrict__ Cf,
    unsigned short* __restrict__ Cbf, int M, int N, int K,
    size_t batchA, size_t batchB, size_t batchC,
    int out_mode, float alpha, int relu) {
    __shared__ __align__(16) unsigned short As[2][64][40];
    __shared__ __align__(16) unsigned short Bs[2][128][40];

    const int tid  = threadIdx.x;
    const int lane = tid & 31;
    const int wave = tid >> 5;
    const int wm   = (wave >> 2) * 32;   // 0,32
    const int wn   = (wave & 3) * 32;    // 0..96
    const int m0   = blockIdx.y * 64;
    const int n0   = blockIdx.x * 128;
    const size_t aB = (size_t)blockIdx.z * batchA;
    const size_t bB = (size_t)blockIdx.z * batchB;
    const size_t cB = (size_t)blockIdx.z * batchC;

    const v8f vz = {0.f,0.f,0.f,0.f,0.f,0.f,0.f,0.f};
    v8f acc[2][2];
    acc[0][0] = vz; acc[0][1] = vz; acc[1][0] = vz; acc[1][1] = vz;

    const int arow = tid >> 2;            // 0..63
    const int acol = (tid & 3) * 8;       // 0,8,16,24

    // hoisted per-thread source pointers (advance by 32 per k-step)
    const unsigned short* aP  = A  + aB + (size_t)(m0 + arow) * K + acol;
    const unsigned short* bP0 = Bt + bB + (size_t)(n0 + arow) * K + acol;
    const unsigned short* bP1 = Bt + bB + (size_t)(n0 + arow + 64) * K + acol;
    const unsigned short* bT  = Bt + bB + (size_t)n0 * K;   // TDM tile origin

#if HAVE_TDM
#define ISSUE_TILE(buf, k0)                                                        \
    do {                                                                           \
        async_ld16(aP + (k0), &As[buf][arow][acol]);                               \
        if (tid == 0) tdm_load_b_tile(bT + (k0), &Bs[buf][0][0], K);               \
    } while (0)
#define A_OPS 1
#define WAIT_TILE(n)  do { WAIT_ASYNC(n); if (wave == 0) WAIT_TENSOR(n); } while (0)
#else
#define ISSUE_TILE(buf, k0)                                                        \
    do {                                                                           \
        async_ld16(aP  + (k0), &As[buf][arow][acol]);                              \
        async_ld16(bP0 + (k0), &Bs[buf][arow][acol]);                              \
        async_ld16(bP1 + (k0), &Bs[buf][arow + 64][acol]);                         \
    } while (0)
#define A_OPS 3
#define WAIT_TILE(n)  WAIT_ASYNC((n) * A_OPS)
#endif

#define COMPUTE(buf)                                                               \
    do {                                                                           \
        v16bf af0 = ldfrag(&As[buf][wm][0],      40, lane);                        \
        v16bf af1 = ldfrag(&As[buf][wm + 16][0], 40, lane);                        \
        v16bf bf0 = ldfrag(&Bs[buf][wn][0],      40, lane);                        \
        v16bf bf1 = ldfrag(&Bs[buf][wn + 16][0], 40, lane);                        \
        acc[0][0] = wmma_bf16(af0, bf0, acc[0][0]);                                \
        acc[0][1] = wmma_bf16(af0, bf1, acc[0][1]);                                \
        acc[1][0] = wmma_bf16(af1, bf0, acc[1][0]);                                \
        acc[1][1] = wmma_bf16(af1, bf1, acc[1][1]);                                \
    } while (0)

    const int nsteps = K >> 5;
    ISSUE_TILE(0, 0);
    int cur = 0;
    for (int s = 0; s < nsteps - 1; ++s) {        // branch-free steady state
        ISSUE_TILE(cur ^ 1, (s + 1) * 32);
        WAIT_TILE(1);                             // previous tile complete
        __syncthreads();
        COMPUTE(cur);
        __syncthreads();
        cur ^= 1;
    }
    WAIT_TILE(0);                                 // last tile
    __syncthreads();
    COMPUTE(cur);

#undef ISSUE_TILE
#undef WAIT_TILE
#undef COMPUTE
#undef A_OPS

    const int half = lane >> 4, c = lane & 15;
    for (int i = 0; i < 2; ++i)
    for (int j = 0; j < 2; ++j)
    for (int t = 0; t < 8; ++t) {
        int row = m0 + wm + i * 16 + t + half * 8;
        int col = n0 + wn + j * 16 + c;
        float v = acc[i][j][t];
        if (bias) v += bias[col];
        v *= alpha;
        if (relu) v = fmaxf(v, 0.f);
        if (out_mode == 0) {
            Cf[cB + (size_t)row * N + col] = v;
            if (Cbf) Cbf[cB + (size_t)row * N + col] = f2bf(v);
        } else if (out_mode == 1) {
            int b = row >> 10, u = row & 1023, h = col >> 6, dh = col & 63;
            Cbf[(((size_t)b * NHEAD + h) * ULEN + u) * DHEAD + dh] = f2bf(v);
        } else if (out_mode == 2) {
            int b = row >> 10, u = row & 1023, h = col >> 6, dh = col & 63;
            Cbf[(((size_t)b * NHEAD + h) * DHEAD + dh) * ULEN + u] = f2bf(v);
        } else {
            Cbf[cB + (size_t)row * N + col] = f2bf(v);
        }
    }
}

// ---------------- flash attention per (b,h,128-query-row block) ----------------
__global__ __launch_bounds__(256) void attn_kernel(
    const unsigned short* __restrict__ Qbf, const unsigned short* __restrict__ Kbf,
    const unsigned short* __restrict__ Vt,  const unsigned short* __restrict__ simbf,
    const float* __restrict__ rel_bias, const int* __restrict__ speakers,
    const unsigned char* __restrict__ mask, const float* __restrict__ gate,
    const float* __restrict__ simscale, unsigned short* __restrict__ Obf) {
    const int b = blockIdx.z, h = blockIdx.y;
    const int tid = threadIdx.x;

    __shared__ float relS[RELMAX + 1];
    __shared__ int   spkS[ULEN];
    __shared__ float mS[ULEN];
    __shared__ __align__(16) unsigned short Pl[8][16][40];

    for (int i = tid; i <= RELMAX; i += 256) relS[i] = rel_bias[h * (RELMAX + 1) + i];
    for (int i = tid; i < ULEN; i += 256) {
        spkS[i] = speakers[b * ULEN + i];
        mS[i]   = mask[b * ULEN + i] ? 1.f : 0.f;
    }
    __syncthreads();

    const int lane = tid & 31, wave = tid >> 5;
    const int half = lane >> 4, c = lane & 15;
    const int u0 = blockIdx.x * 128 + wave * 16;

    const size_t hb = (((size_t)b * NHEAD) + h) * ULEN * DHEAD;
    const unsigned short* Qp = Qbf + hb;   // [u][dh], pre-scaled by 1/sqrt(Dh)
    const unsigned short* Kp = Kbf + hb;   // [u][dh]
    const unsigned short* Vp = Vt  + hb;   // [dh][u]

    v16bf qf0 = ldfrag(Qp + (size_t)u0 * DHEAD,      DHEAD, lane);
    v16bf qf1 = ldfrag(Qp + (size_t)u0 * DHEAD + 32, DHEAD, lane);

    const v8f vz = {0.f,0.f,0.f,0.f,0.f,0.f,0.f,0.f};
    v8f oacc[4]; oacc[0] = vz; oacc[1] = vz; oacc[2] = vz; oacc[3] = vz;
    float mrow[8], lrow[8];
    for (int j = 0; j < 8; ++j) { mrow[j] = -1e30f; lrow[j] = 0.f; }

    const float gh  = gate[h];
    const float ssh = simscale[h];

    for (int w0 = 0; w0 < ULEN; w0 += 32) {
        if (w0 + 32 < ULEN) {   // prefetch next K/V tiles into cache
            __builtin_prefetch(Kp + (size_t)(w0 + 32) * DHEAD + (lane & 15) * DHEAD, 0, 1);
            __builtin_prefetch(Vp + (size_t)(lane & 15) * ULEN + w0 + 32, 0, 1);
        }
        // ---- scores S(16x32) = q @ k^T ----
        v8f sacc[2]; sacc[0] = vz; sacc[1] = vz;
        for (int t = 0; t < 2; ++t) {
            v16bf kf0 = ldfrag(Kp + (size_t)(w0 + t * 16) * DHEAD,      DHEAD, lane);
            v16bf kf1 = ldfrag(Kp + (size_t)(w0 + t * 16) * DHEAD + 32, DHEAD, lane);
            sacc[t] = wmma_bf16(qf0, kf0, sacc[t]);
            sacc[t] = wmma_bf16(qf1, kf1, sacc[t]);
        }
        // ---- bias terms + per-row online softmax ----
        float pel[2][8];
        for (int j = 0; j < 8; ++j) {
            const int u = u0 + j + half * 8;
            float best = -1e30f;
            for (int t = 0; t < 2; ++t) {
                const int w = w0 + t * 16 + c;
                float e = sacc[t][j];
                int dist = u - w; if (dist < 0) dist = -dist;
                if (dist > RELMAX) dist = RELMAX;
                e += relS[dist];
                e += gh * ((spkS[u] != spkS[w]) ? 1.f : 0.f);
                float sim = bf2f(simbf[((size_t)b * ULEN + u) * ULEN + w]);
                e += ssh * sim * mS[u] * mS[w];
                e = (mS[w] > 0.f) ? e : -1e30f;
                pel[t][j] = e;
                best = fmaxf(best, e);
            }
            float v = best;
            for (int off = 1; off < 16; off <<= 1) v = fmaxf(v, __shfl_xor(v, off, 32));
            float mnew = fmaxf(mrow[j], v);
            float corr = __expf(mrow[j] - mnew);
            mrow[j] = mnew;
            float psum = 0.f;
            for (int t = 0; t < 2; ++t) {
                float p = __expf(pel[t][j] - mnew);
                pel[t][j] = p;
                psum += p;
            }
            for (int off = 1; off < 16; off <<= 1) psum += __shfl_xor(psum, off, 32);
            lrow[j] = lrow[j] * corr + psum;
            oacc[0][j] *= corr; oacc[1][j] *= corr;
            oacc[2][j] *= corr; oacc[3][j] *= corr;
        }
        // ---- P (C-layout) -> LDS -> A fragment ----
        for (int j = 0; j < 8; ++j) {
            const int r = j + half * 8;
            Pl[wave][r][c]      = f2bf(pel[0][j]);
            Pl[wave][r][16 + c] = f2bf(pel[1][j]);
        }
        v16bf pf = ldfrag(&Pl[wave][0][0], 40, lane);
        // ---- O += P @ V  (V^T layout gives contiguous B fragments) ----
        for (int t = 0; t < 4; ++t) {
            v16bf vf = ldfrag(Vp + (size_t)(t * 16) * ULEN + w0, ULEN, lane);
            oacc[t] = wmma_bf16(pf, vf, oacc[t]);
        }
    }

    for (int j = 0; j < 8; ++j) {
        const int u = u0 + j + half * 8;
        const float inv = (lrow[j] > 0.f) ? 1.f / lrow[j] : 0.f;
        const float mq  = mS[u];
        for (int t = 0; t < 4; ++t) {
            const int dh = t * 16 + c;
            Obf[((size_t)(b * ULEN + u)) * DMODEL + h * DHEAD + dh] =
                f2bf(oacc[t][j] * inv * mq);
        }
    }
}

// ---------------- residual + LayerNorm (f32 out, optional bf16 out) ----------------
__global__ __launch_bounds__(256) void add_ln_kernel(
    const float* __restrict__ X, const float* __restrict__ Y,
    const float* __restrict__ g, const float* __restrict__ be,
    const unsigned char* __restrict__ mask,
    float* __restrict__ outF, unsigned short* __restrict__ outBf) {
    const int row = blockIdx.x;
    const int tid = threadIdx.x;
    __shared__ float red[256];
    const float* xr = X + (size_t)row * DMODEL;
    const float* yr = Y + (size_t)row * DMODEL;
    float v0 = xr[tid] + yr[tid];
    float v1 = xr[tid + 256] + yr[tid + 256];
    red[tid] = v0 + v1;
    __syncthreads();
    for (int o = 128; o > 0; o >>= 1) {
        if (tid < o) red[tid] += red[tid + o];
        __syncthreads();
    }
    float mu = red[0] * (1.f / DMODEL);
    __syncthreads();
    float d0 = v0 - mu, d1 = v1 - mu;
    red[tid] = d0 * d0 + d1 * d1;
    __syncthreads();
    for (int o = 128; o > 0; o >>= 1) {
        if (tid < o) red[tid] += red[tid + o];
        __syncthreads();
    }
    float rs = rsqrtf(red[0] * (1.f / DMODEL) + 1e-5f);
    float mk = mask[row] ? 1.f : 0.f;
    float o0 = (d0 * rs * g[tid] + be[tid]) * mk;
    float o1 = (d1 * rs * g[tid + 256] + be[tid + 256]) * mk;
    outF[(size_t)row * DMODEL + tid]       = o0;
    outF[(size_t)row * DMODEL + tid + 256] = o1;
    if (outBf) {
        outBf[(size_t)row * DMODEL + tid]       = f2bf(o0);
        outBf[(size_t)row * DMODEL + tid + 256] = f2bf(o1);
    }
}

// ---------------- host side ----------------
extern "C" void kernel_launch(void* const* d_in, const int* in_sizes, int n_in,
                              void* d_out, int out_size, void* d_ws, size_t ws_size,
                              hipStream_t stream) {
    const float* X      = (const float*)d_in[0];
    const unsigned char* mask = (const unsigned char*)d_in[1];
    const int*   spk    = (const int*)d_in[2];
    const float* Wq = (const float*)d_in[3],  *bq = (const float*)d_in[4];
    const float* Wk = (const float*)d_in[5],  *bk = (const float*)d_in[6];
    const float* Wv = (const float*)d_in[7],  *bv = (const float*)d_in[8];
    const float* Wo = (const float*)d_in[9],  *bo = (const float*)d_in[10];
    const float* rel = (const float*)d_in[11];
    const float* gate = (const float*)d_in[12];
    const float* sscale = (const float*)d_in[13];
    const float* g1 = (const float*)d_in[14], *be1 = (const float*)d_in[15];
    const float* g2 = (const float*)d_in[16], *be2 = (const float*)d_in[17];
    const float* W1 = (const float*)d_in[18], *bf1 = (const float*)d_in[19];
    const float* W2 = (const float*)d_in[20], *bf2 = (const float*)d_in[21];
    float* out = (float*)d_out;

    // workspace layout (bytes)
    char* w = (char*)d_ws;
    size_t off = 0;
    auto take = [&](size_t bytes) { char* p = w + off; off += (bytes + 255) & ~(size_t)255; return p; };
    unsigned short* Xbf   = (unsigned short*)take((size_t)NROWS * DMODEL * 2);
    unsigned short* Xnbf  = (unsigned short*)take((size_t)NROWS * DMODEL * 2);
    unsigned short* WqT   = (unsigned short*)take((size_t)DMODEL * DMODEL * 2);
    unsigned short* WkT   = (unsigned short*)take((size_t)DMODEL * DMODEL * 2);
    unsigned short* WvT   = (unsigned short*)take((size_t)DMODEL * DMODEL * 2);
    unsigned short* WoT   = (unsigned short*)take((size_t)DMODEL * DMODEL * 2);
    unsigned short* W1T   = (unsigned short*)take((size_t)DMODEL * DFF * 2);
    unsigned short* W2T   = (unsigned short*)take((size_t)DFF * DMODEL * 2);
    unsigned short* Qb    = (unsigned short*)take((size_t)NROWS * DMODEL * 2);
    unsigned short* Kb    = (unsigned short*)take((size_t)NROWS * DMODEL * 2);
    unsigned short* Vtb   = (unsigned short*)take((size_t)NROWS * DMODEL * 2);
    unsigned short* simbf = (unsigned short*)take((size_t)BATCH * ULEN * ULEN * 2);
    unsigned short* Attbf = (unsigned short*)take((size_t)NROWS * DMODEL * 2);
    float*          ProjF = (float*)take((size_t)NROWS * DMODEL * 4);
    float*          X1F   = (float*)take((size_t)NROWS * DMODEL * 4);
    unsigned short* X1bf  = (unsigned short*)take((size_t)NROWS * DMODEL * 2);
    unsigned short* Hbf   = (unsigned short*)take((size_t)NROWS * DFF * 2);
    float*          FfnF  = (float*)take((size_t)NROWS * DMODEL * 4);

    // 1) X -> bf16 + normalized bf16
    prep_x_kernel<<<NROWS, 256, 0, stream>>>(X, Xbf, Xnbf);

    // 2) weight transposes (N x K bf16)
    wtrans_kernel<<<dim3(DMODEL/16, DMODEL/16), dim3(16,16), 0, stream>>>(Wq, WqT, DMODEL, DMODEL);
    wtrans_kernel<<<dim3(DMODEL/16, DMODEL/16), dim3(16,16), 0, stream>>>(Wk, WkT, DMODEL, DMODEL);
    wtrans_kernel<<<dim3(DMODEL/16, DMODEL/16), dim3(16,16), 0, stream>>>(Wv, WvT, DMODEL, DMODEL);
    wtrans_kernel<<<dim3(DMODEL/16, DMODEL/16), dim3(16,16), 0, stream>>>(Wo, WoT, DMODEL, DMODEL);
    wtrans_kernel<<<dim3(DFF/16,    DMODEL/16), dim3(16,16), 0, stream>>>(W1, W1T, DMODEL, DFF);
    wtrans_kernel<<<dim3(DMODEL/16, DFF/16),    dim3(16,16), 0, stream>>>(W2, W2T, DFF, DMODEL);

    // 3) Q,K,V projections (Q pre-scaled by 1/sqrt(Dh)); V stored transposed
    gemm_bf16_kernel<<<dim3(DMODEL/128, NROWS/64, 1), 256, 0, stream>>>(
        Xbf, WqT, bq, nullptr, Qb,  NROWS, DMODEL, DMODEL, 0,0,0, 1, 0.125f, 0);
    gemm_bf16_kernel<<<dim3(DMODEL/128, NROWS/64, 1), 256, 0, stream>>>(
        Xbf, WkT, bk, nullptr, Kb,  NROWS, DMODEL, DMODEL, 0,0,0, 1, 1.0f, 0);
    gemm_bf16_kernel<<<dim3(DMODEL/128, NROWS/64, 1), 256, 0, stream>>>(
        Xbf, WvT, bv, nullptr, Vtb, NROWS, DMODEL, DMODEL, 0,0,0, 2, 1.0f, 0);

    // 4) cosine-sim matrix per batch: sim = Xn @ Xn^T (bf16 out)
    gemm_bf16_kernel<<<dim3(ULEN/128, ULEN/64, BATCH), 256, 0, stream>>>(
        Xnbf, Xnbf, nullptr, nullptr, simbf, ULEN, ULEN, DMODEL,
        (size_t)ULEN*DMODEL, (size_t)ULEN*DMODEL, (size_t)ULEN*ULEN, 3, 1.0f, 0);

    // 5) flash attention with rel-bias / speaker-gate / sim terms
    attn_kernel<<<dim3(ULEN/128, NHEAD, BATCH), 256, 0, stream>>>(
        Qb, Kb, Vtb, simbf, rel, spk, mask, gate, sscale, Attbf);

    // 6) output projection
    gemm_bf16_kernel<<<dim3(DMODEL/128, NROWS/64, 1), 256, 0, stream>>>(
        Attbf, WoT, bo, ProjF, nullptr, NROWS, DMODEL, DMODEL, 0,0,0, 0, 1.0f, 0);

    // 7) X1 = LN(X + proj) * mask  (f32 + bf16)
    add_ln_kernel<<<NROWS, 256, 0, stream>>>(X, ProjF, g1, be1, mask, X1F, X1bf);

    // 8) FFN layer 1 with ReLU -> bf16 activations
    gemm_bf16_kernel<<<dim3(DFF/128, NROWS/64, 1), 256, 0, stream>>>(
        X1bf, W1T, bf1, nullptr, Hbf, NROWS, DFF, DMODEL, 0,0,0, 3, 1.0f, 1);

    // 9) FFN layer 2 -> f32
    gemm_bf16_kernel<<<dim3(DMODEL/128, NROWS/64, 1), 256, 0, stream>>>(
        Hbf, W2T, bf2, FfnF, nullptr, NROWS, DMODEL, DFF, 0,0,0, 0, 1.0f, 0);

    // 10) X2 = LN(X1 + ffn) * mask -> d_out
    add_ln_kernel<<<NROWS, 256, 0, stream>>>(X1F, FfnF, g2, be2, mask, out, nullptr);

    (void)in_sizes; (void)n_in; (void)out_size; (void)ws_size;
}
